// SAGEConvWithEdgeAttr_24051816857689
// MI455X (gfx1250) — compile-verified, matrix operand order checked by reading, not run on hardware
//
#include <hip/hip_runtime.h>
#include <hip/hip_bf16.h>

#define N_NODES 100000
#define N_EDGESC 1600000
#define IN_C 128
#define OUT_C 64
#define EDGE_DIM 32
#define BN_EPS 1e-5f

typedef __attribute__((ext_vector_type(16))) _Float16 v16h;
typedef __attribute__((ext_vector_type(8)))  float    v8f;

// ---- WMMA helpers (CDNA5 wave32, 16x16x32 f16 -> f32) ----

__device__ __forceinline__ int kidx(int j, int half) {
  // 16-bit A/B layout: lanes 0-15 hold K = 0..7 (j<8) and 16..23 (j>=8);
  // lanes 16-31 hold K = 8..15 and 24..31.
  return (j < 8) ? (half * 8 + j) : (16 + half * 8 + (j - 8));
}

__device__ __forceinline__ v8f wmma16(v16h a, v16h b, v8f c) {
  return __builtin_amdgcn_wmma_f32_16x16x32_f16(
      /*neg_a=*/false, a, /*neg_b=*/false, b,
      /*c_mod=*/(short)0, c, /*reuse_a=*/false, /*reuse_b=*/false);
}

__device__ __forceinline__ v8f splat8(float v) {
  v8f t;
#pragma unroll
  for (int j = 0; j < 8; ++j) t[j] = v;
  return t;
}

// Build an A-tile (16x32 f16) from an f32 row: two contiguous 8-float chunks.
__device__ __forceinline__ v16h load_a_f32(const float* rowp, int kbase, int half) {
  v16h a;
  const float* p0 = rowp + kbase + half * 8;
  const float* p1 = p0 + 16;
#pragma unroll
  for (int j = 0; j < 8; ++j) { a[j] = (_Float16)p0[j]; a[8 + j] = (_Float16)p1[j]; }
  return a;
}

// Load a pre-packed B tile: one contiguous 32B chunk per lane.
__device__ __forceinline__ v16h load_b_pk(const _Float16* pk, int kstep, int nt, int lane) {
  return *(const v16h*)(pk + (((kstep * 4 + nt) * 32 + lane) * 16));
}

// ---- prep: fused weights + pack B matrices into WMMA layout ----

__device__ void pack_weight(_Float16* __restrict__ dst, const float* __restrict__ src,
                            int ksteps, int tid, int nthreads) {
  int total = ksteps * 4 * 512;
  for (int idx = tid; idx < total; idx += nthreads) {
    int kstep = idx / 2048;
    int rem   = idx % 2048;
    int nt    = rem / 512;
    int rem2  = rem % 512;
    int lane  = rem2 / 16;
    int j     = rem2 % 16;
    int n = lane & 15, half = lane >> 4;
    int k = kstep * 32 + kidx(j, half);
    dst[idx] = (_Float16)src[k * OUT_C + nt * 16 + n];
  }
}

__global__ void prep_kernel(const float* __restrict__ W_l, const float* __restrict__ W_r,
                            const float* __restrict__ W_e, const float* __restrict__ W_g,
                            const float* __restrict__ b_e, const float* __restrict__ b_g,
                            float* __restrict__ we2f, float* __restrict__ bg2,
                            _Float16* __restrict__ pk) {
  int tid = threadIdx.x;
  // We2 = W_e @ Wg2 (Wg2 = rows 64..127 of W_g)
  for (int i = tid; i < EDGE_DIM * OUT_C; i += blockDim.x) {
    int r = i >> 6, c = i & 63;
    float s = 0.f;
    for (int k = 0; k < OUT_C; ++k) s += W_e[r * OUT_C + k] * W_g[(OUT_C + k) * OUT_C + c];
    we2f[i] = s;
  }
  // bg2 = b_g + b_e @ Wg2
  if (tid < OUT_C) {
    float s = b_g[tid];
    for (int k = 0; k < OUT_C; ++k) s += b_e[k] * W_g[(OUT_C + k) * OUT_C + tid];
    bg2[tid] = s;
  }
  __syncthreads();
  pack_weight(pk + 0,     W_l,  4, tid, blockDim.x);   // 128x64
  pack_weight(pk + 8192,  W_r,  4, tid, blockDim.x);   // 128x64
  pack_weight(pk + 16384, W_g,  2, tid, blockDim.x);   // Wg1 = rows 0..63
  pack_weight(pk + 20480, W_e,  1, tid, blockDim.x);   // 32x64
  pack_weight(pk + 22528, we2f, 1, tid, blockDim.x);   // 32x64
}

__global__ void zero_kernel(float* __restrict__ p, int n) {
  int i = blockIdx.x * blockDim.x + threadIdx.x;
  if (i < n) p[i] = 0.f;
}

// ---- hl = x @ W_l  (one wave per 16-node tile) ----

__global__ void node_gemm_kernel(const float* __restrict__ x,
                                 const _Float16* __restrict__ pkW,
                                 float* __restrict__ out, int ntiles) {
  int wave = blockIdx.x * (blockDim.x >> 5) + (threadIdx.x >> 5);
  if (wave >= ntiles) return;                 // wave-uniform: EXEC stays all-1s
  int lane = threadIdx.x & 31;
  int m = lane & 15, half = lane >> 4;
  int m0 = wave * 16;
  v8f acc[4];
#pragma unroll
  for (int nt = 0; nt < 4; ++nt) acc[nt] = splat8(0.f);
  const float* rowp = x + (size_t)(m0 + m) * IN_C;
#pragma unroll
  for (int ks = 0; ks < 4; ++ks) {
    v16h a = load_a_f32(rowp, ks * 32, half);
#pragma unroll
    for (int nt = 0; nt < 4; ++nt)
      acc[nt] = wmma16(a, load_b_pk(pkW, ks, nt, lane), acc[nt]);
  }
#pragma unroll
  for (int nt = 0; nt < 4; ++nt)
#pragma unroll
    for (int r = 0; r < 8; ++r)
      out[(size_t)(m0 + r + 8 * half) * OUT_C + nt * 16 + m] = acc[nt][r];
}

// ---- scatter-mean numerator/denominator: agg[col] += hl[row]; cnt[col] += 1 ----

__global__ void scatter_kernel(const float* __restrict__ hl,
                               const int* __restrict__ row, const int* __restrict__ col,
                               float* __restrict__ agg, float* __restrict__ cnt) {
  int gid = blockIdx.x * blockDim.x + threadIdx.x;  // E*16 threads, 4 ch each
  int e = gid >> 4;
  if (e >= N_EDGESC) return;
  int c4 = (gid & 15) * 4;
  int r = row[e], co = col[e];
  const float4 v = *(const float4*)(hl + (size_t)r * OUT_C + c4);
  float* dst = agg + (size_t)co * OUT_C + c4;
  atomicAdd(dst + 0, v.x);
  atomicAdd(dst + 1, v.y);
  atomicAdd(dst + 2, v.z);
  atomicAdd(dst + 3, v.w);
  if ((gid & 15) == 0) atomicAdd(cnt + co, 1.0f);
}

// ---- out_pre = agg/max(cnt,1) + b_l + x@W_r (in place into agg); zero contrib ----

__global__ void combine_kernel(const float* __restrict__ x,
                               const _Float16* __restrict__ pkWr,
                               float* __restrict__ agg, const float* __restrict__ cnt,
                               const float* __restrict__ b_l,
                               float* __restrict__ contrib, int ntiles) {
  int wave = blockIdx.x * (blockDim.x >> 5) + (threadIdx.x >> 5);
  if (wave >= ntiles) return;
  int lane = threadIdx.x & 31;
  int m = lane & 15, half = lane >> 4;
  int m0 = wave * 16;
  v8f acc[4];
#pragma unroll
  for (int nt = 0; nt < 4; ++nt) acc[nt] = splat8(0.f);
  const float* rowp = x + (size_t)(m0 + m) * IN_C;
#pragma unroll
  for (int ks = 0; ks < 4; ++ks) {
    v16h a = load_a_f32(rowp, ks * 32, half);
#pragma unroll
    for (int nt = 0; nt < 4; ++nt)
      acc[nt] = wmma16(a, load_b_pk(pkWr, ks, nt, lane), acc[nt]);
  }
#pragma unroll
  for (int r = 0; r < 8; ++r) {
    int i = m0 + r + 8 * half;
    float inv = 1.0f / fmaxf(cnt[i], 1.0f);
#pragma unroll
    for (int nt = 0; nt < 4; ++nt) {
      size_t idx = (size_t)i * OUT_C + nt * 16 + m;
      agg[idx] = agg[idx] * inv + b_l[nt * 16 + m] + acc[nt][r];
      contrib[idx] = 0.0f;
    }
  }
}

// ---- fused edge gating: gate = sigmoid(out[col]@Wg1 + edge_attr@We2 + bg2);
//      contrib[col] += gate * (edge_attr@W_e + b_e)  (one wave per 16-edge tile) ----

__global__ void gate_kernel(const float* __restrict__ outp,
                            const float* __restrict__ eattr,
                            const int* __restrict__ col,
                            const _Float16* __restrict__ pk,
                            const float* __restrict__ bg2,
                            const float* __restrict__ b_e,
                            float* __restrict__ contrib, int etiles) {
  int wave = blockIdx.x * (blockDim.x >> 5) + (threadIdx.x >> 5);
  if (wave >= etiles) return;
  int lane = threadIdx.x & 31;
  int m = lane & 15, half = lane >> 4;
  int e0 = wave * 16;
  const _Float16* pkWg1 = pk + 16384;
  const _Float16* pkWe  = pk + 20480;
  const _Float16* pkWe2 = pk + 22528;

  v8f ea_acc[4], g_acc[4];
#pragma unroll
  for (int nt = 0; nt < 4; ++nt) {
    ea_acc[nt] = splat8(b_e[nt * 16 + m]);
    g_acc[nt]  = splat8(bg2[nt * 16 + m]);
  }
  {
    v16h a = load_a_f32(eattr + (size_t)(e0 + m) * EDGE_DIM, 0, half);
#pragma unroll
    for (int nt = 0; nt < 4; ++nt) {
      ea_acc[nt] = wmma16(a, load_b_pk(pkWe,  0, nt, lane), ea_acc[nt]);
      g_acc[nt]  = wmma16(a, load_b_pk(pkWe2, 0, nt, lane), g_acc[nt]);
    }
  }
  int cm = col[e0 + m];
  const float* rowp = outp + (size_t)cm * OUT_C;
#pragma unroll
  for (int ks = 0; ks < 2; ++ks) {
    v16h a = load_a_f32(rowp, ks * 32, half);
#pragma unroll
    for (int nt = 0; nt < 4; ++nt)
      g_acc[nt] = wmma16(a, load_b_pk(pkWg1, ks, nt, lane), g_acc[nt]);
  }
  int colv[8];
#pragma unroll
  for (int r = 0; r < 8; ++r) colv[r] = col[e0 + 8 * half + r];
#pragma unroll
  for (int nt = 0; nt < 4; ++nt)
#pragma unroll
    for (int r = 0; r < 8; ++r) {
      float g = 1.0f / (1.0f + __expf(-g_acc[nt][r]));
      atomicAdd(contrib + (size_t)colv[r] * OUT_C + nt * 16 + m, g * ea_acc[nt][r]);
    }
}

// ---- BatchNorm stats + apply ----

__global__ void bn_stats_kernel(const float* __restrict__ outp, const float* __restrict__ contrib,
                                float* __restrict__ stats, int total) {
  __shared__ float sh[2 * OUT_C];
  int t = threadIdx.x;
  if (t < 2 * OUT_C) sh[t] = 0.f;
  __syncthreads();
  int gtid = blockIdx.x * blockDim.x + t;
  int stride = gridDim.x * blockDim.x;  // multiple of 64 -> channel constant per thread
  int ch = gtid & 63;
  float s = 0.f, s2 = 0.f;
  for (int i = gtid; i < total; i += stride) {
    float v = outp[i] + contrib[i];
    s += v; s2 += v * v;
  }
  atomicAdd(&sh[ch], s);
  atomicAdd(&sh[OUT_C + ch], s2);
  __syncthreads();
  if (t < 2 * OUT_C) atomicAdd(&stats[t], sh[t]);
}

__global__ void bn_apply_kernel(const float* __restrict__ outp, const float* __restrict__ contrib,
                                const float* __restrict__ stats,
                                const float* __restrict__ gamma, const float* __restrict__ beta,
                                float* __restrict__ out, int total) {
  int i = blockIdx.x * blockDim.x + threadIdx.x;
  if (i >= total) return;
  int ch = i & 63;
  float invN = 1.0f / (float)N_NODES;
  float mu  = stats[ch] * invN;
  float var = stats[OUT_C + ch] * invN - mu * mu;
  float v = outp[i] + contrib[i];
  float r = gamma[ch] * (v - mu) * rsqrtf(var + BN_EPS) + beta[ch];
  out[i] = fmaxf(2.0f * r, 0.0f);
}

extern "C" void kernel_launch(void* const* d_in, const int* in_sizes, int n_in,
                              void* d_out, int out_size, void* d_ws, size_t ws_size,
                              hipStream_t stream) {
  (void)in_sizes; (void)n_in; (void)out_size; (void)ws_size;
  const float* x     = (const float*)d_in[0];
  const int*   ei    = (const int*)d_in[1];   // int32 (JAX x64 disabled)
  const float* eattr = (const float*)d_in[2];
  const float* W_l   = (const float*)d_in[3];
  const float* b_l   = (const float*)d_in[4];
  const float* W_r   = (const float*)d_in[5];
  const float* W_e   = (const float*)d_in[6];
  const float* b_e   = (const float*)d_in[7];
  const float* W_g   = (const float*)d_in[8];
  const float* b_g   = (const float*)d_in[9];
  const float* gamma = (const float*)d_in[10];
  const float* beta  = (const float*)d_in[11];

  char* ws = (char*)d_ws;
  float*    hl    = (float*)(ws + 0);            // N*64; later reused as contrib
  float*    agg   = (float*)(ws + 25600000);     // N*64; becomes out_pre in place
  float*    cnt   = (float*)(ws + 51200000);     // N
  float*    stats = (float*)(ws + 51600128);     // 128 (sum, sumsq)
  float*    we2f  = (float*)(ws + 51600640);     // 32*64
  float*    bg2   = (float*)(ws + 51608832);     // 64
  _Float16* pk    = (_Float16*)(ws + 51609088);  // 24576 packed f16 weights

  const int* row = ei;
  const int* col = ei + N_EDGESC;

  // zero agg + cnt + pad + stats (contiguous region)
  int zn = (51600128 + 512 - 25600000) / 4;
  zero_kernel<<<(zn + 255) / 256, 256, 0, stream>>>(agg, zn);

  prep_kernel<<<1, 256, 0, stream>>>(W_l, W_r, W_e, W_g, b_e, b_g, we2f, bg2, pk);

  int ntile = N_NODES / 16;          // 6250
  int nblk  = (ntile + 7) / 8;       // 8 waves / 256-thread block
  node_gemm_kernel<<<nblk, 256, 0, stream>>>(x, pk, hl, ntile);

  scatter_kernel<<<(N_EDGESC * 16) / 256, 256, 0, stream>>>(hl, row, col, agg, cnt);

  combine_kernel<<<nblk, 256, 0, stream>>>(x, pk + 8192, agg, cnt, b_l, hl, ntile);

  int etile = N_EDGESC / 16;         // 100000
  gate_kernel<<<(etile + 7) / 8, 256, 0, stream>>>(agg, eattr, col, pk, bg2, b_e, hl, etile);

  bn_stats_kernel<<<512, 256, 0, stream>>>(agg, hl, stats, N_NODES * OUT_C);
  bn_apply_kernel<<<(N_NODES * OUT_C + 255) / 256, 256, 0, stream>>>(
      agg, hl, stats, gamma, beta, (float*)d_out, N_NODES * OUT_C);
}
